// YOLOXHead_37898791419965
// MI455X (gfx1250) — compile-verified, matrix operand order checked by reading, not run on hardware
//
#include <hip/hip_runtime.h>

// ---------------------------------------------------------------------------
// MI455X (gfx1250, wave32) implementation of the YOLOX head + MSA reference.
// All GEMM-shaped math (implicit-GEMM convs, QKV, attention, MLPs) runs on
// v_wmma_f32_16x16x32_f16 (f16 inputs, f32 accumulate). Scalar decision logic
// (BN/SiLU, sigmoid, box decode, top-k, NMS, softmax, norms) stays in f32.
//
// GEMM kernel: 128 threads (4 waves), 64x64 macro-tile, each wave computes a
// 32x32 quadrant via 2x2 accumulators -> 4 v_wmma per staged K-slice of 32.
// Tiles are staged in LDS directly in ISA 7.12.2 fragment order so each lane
// reads its v16h fragment as two contiguous ds_load_b128.
//
// d_in layout assumption (setup_inputs insertion order, params flattened
// depth-first in insertion order):
//   0 x0, 1 x1, 2 x2, 3 other_cls_feat, 4 other_reg_feat,
//   5 other_cls_scores, 6 other_reg_scores,
//   7 + 41*l : level l = [stem(w,g,b,m,v), cls0, cls1, c20, c21, reg0, reg1
//                         (5 leaves each), cls_w, cls_b, regp_w, regp_b,
//                         obj_w, obj_b]
//   130 wqkv_cls, 131 wqkv_reg, 132 w1, 133 b1, 134 w2, 135 b2, 136 wp, 137 bp
// ---------------------------------------------------------------------------

typedef _Float16 v16h __attribute__((ext_vector_type(16)));
typedef float    v8f  __attribute__((ext_vector_type(8)));

#define NT    20160   // total anchors (96*160 + 48*80 + 24*40)
#define PRE   750
#define NTOK  600     // 30 + 570 tokens in attention

// ===========================================================================
// WMMA GEMM: D[M,N] = act(A[M,K] * B[K,N])
//  - A: f16; plain (row stride ldA) or implicit-im2col 3x3 SAME (Cin==256)
//  - B: f16 packed [K][N] row-major
//  - epilogue: act 0=none, 1=BN(scale,bias)+SiLU, 2=bias+sigmoid, 3=bias
//  - batched over blockIdx.z with element strides bA/bB/bC
// ===========================================================================
__global__ __launch_bounds__(128)
void gemm_wmma(const _Float16* __restrict__ A, int ldA,
               const _Float16* __restrict__ B,
               float* __restrict__ outF, _Float16* __restrict__ outH, int ldC,
               const float* __restrict__ scl, const float* __restrict__ bia, int act,
               int M, int N, int K,
               int im2, int H, int W, int Cin,
               long long bA, long long bB, long long bC)
{
    // fragment-ordered staging: [subtile][lane][slot], row stride 32 halves
    // (64B) so each lane's 16-half fragment is 32B-aligned contiguous memory.
    __shared__ __align__(32) _Float16 AF[4][32][32];
    __shared__ __align__(32) _Float16 BF[4][32][32];

    const int tm = blockIdx.x * 64, tn = blockIdx.y * 64;
    const int z  = blockIdx.z;
    A += (long long)z * bA;
    B += (long long)z * bB;
    const long long co = (long long)z * bC;
    const int tid = threadIdx.x;
    (void)Cin;

    // A loader: thread owns row ar = tid>>1, cols [(tid&1)*16, +16)
    const int ar   = tid >> 1;
    const int ac0  = (tid & 1) * 16;
    const int am   = tm + ar;
    const bool amok = (am < M);
    int ay = 0, ax = 0;
    if (im2 && amok) { ay = am / W; ax = am - ay * W; }   // hoisted: loop-invariant
    const int ami = ar >> 4, arr = ar & 15;

    // B loader: thread owns row br = tid&31, cols [(tid>>5)*16, +16)
    const int br  = tid & 31;
    const int bc0 = (tid >> 5) * 16;
    const int bfl0 = ((br >> 3) & 1) * 16;
    const int bfs  = (br & 7) + ((br >> 4) & 1) * 8;

    // compute assignment: wave wv owns 32x32 quadrant (miB..miB+1, niB..niB+1)
    const int wv = tid >> 5, lane = tid & 31;
    const int miB = (wv & 1) * 2, niB = (wv >> 1) * 2;

    v8f zero = {0.f, 0.f, 0.f, 0.f, 0.f, 0.f, 0.f, 0.f};
    v8f acc00 = zero, acc01 = zero, acc10 = zero, acc11 = zero;

    for (int k0 = 0; k0 < K; k0 += 32) {
        // gfx1250 prefetch of next B K-slice -> global_prefetch_b8
        if (k0 + 32 < K)
            __builtin_prefetch(B + (long long)(k0 + 32) * N + tn, 0, 1);

        // ---- stage A (64 rows x 32 K) in fragment order ----
        #pragma unroll
        for (int i = 0; i < 16; i++) {
            int c = ac0 + i, kk = k0 + c;
            _Float16 val = (_Float16)0.0f;
            if (amok && kk < K) {
                if (im2) {               // Cin == 256 on every im2col path
                    int t9 = kk >> 8, ic = kk & 255;
                    int dy = (t9 >= 6) ? 2 : ((t9 >= 3) ? 1 : 0);
                    int dx = t9 - dy * 3;
                    int py = ay + dy - 1, px = ax + dx - 1;
                    if (py >= 0 && py < H && px >= 0 && px < W)
                        val = A[(long long)(py * W + px) * 256 + ic];
                } else {
                    val = A[(long long)am * ldA + kk];
                }
            }
            int fl = ((c >> 3) & 1) * 16 + arr;          // lane within subtile
            int fs = (c & 7) + ((c >> 4) & 1) * 8;       // slot within fragment
            AF[ami][fl][fs] = val;
        }
        // ---- stage B (32 K x 64 cols) in fragment order ----
        {
            const int kk = k0 + br;
            #pragma unroll
            for (int i = 0; i < 16; i++) {
                int cb = bc0 + i, nn = tn + cb;
                _Float16 val = (_Float16)0.0f;
                if (kk < K && nn < N) val = B[(long long)kk * N + nn];
                BF[cb >> 4][bfl0 + (cb & 15)][bfs] = val;
            }
        }
        __syncthreads();

        // two ds_load_b128 per fragment
        v16h a0 = *(const v16h*)&AF[miB    ][lane][0];
        v16h a1 = *(const v16h*)&AF[miB + 1][lane][0];
        v16h b0 = *(const v16h*)&BF[niB    ][lane][0];
        v16h b1 = *(const v16h*)&BF[niB + 1][lane][0];
        acc00 = __builtin_amdgcn_wmma_f32_16x16x32_f16(false, a0, false, b0, (short)0, acc00, false, false);
        acc01 = __builtin_amdgcn_wmma_f32_16x16x32_f16(false, a0, false, b1, (short)0, acc01, false, false);
        acc10 = __builtin_amdgcn_wmma_f32_16x16x32_f16(false, a1, false, b0, (short)0, acc10, false, false);
        acc11 = __builtin_amdgcn_wmma_f32_16x16x32_f16(false, a1, false, b1, (short)0, acc11, false, false);
        __syncthreads();
    }

    // epilogue: C/D layout = VGPR e: lanes0-15 M=e, lanes16-31 M=8+e; N=lane&15
    #pragma unroll
    for (int i = 0; i < 2; i++) {
        #pragma unroll
        for (int j = 0; j < 2; j++) {
            v8f acc = (i == 0) ? ((j == 0) ? acc00 : acc01)
                               : ((j == 0) ? acc10 : acc11);
            int n  = tn + (niB + j) * 16 + (lane & 15);
            int m0 = tm + (miB + i) * 16 + (lane >> 4) * 8;
            if (n >= N) continue;
            float sc = scl ? scl[n] : 1.0f;
            float bi = bia ? bia[n] : 0.0f;
            for (int e = 0; e < 8; e++) {
                int m = m0 + e;
                if (m >= M) continue;
                float d = acc[e];
                if (act == 1)      { d = d * sc + bi; d = d * (1.0f / (1.0f + __expf(-d))); }
                else if (act == 2) { d += bi; d = 1.0f / (1.0f + __expf(-d)); }
                else if (act == 3) { d += bi; }
                if (outF) outF[co + (long long)m * ldC + n] = d;
                if (outH) outH[co + (long long)m * ldC + n] = (_Float16)d;
            }
        }
    }
}

// ===========================================================================
// Small helper kernels (data prep, f32 scalar logic)
// ===========================================================================
__global__ void f32_to_h_kernel(const float* __restrict__ s, _Float16* __restrict__ d, int n) {
    int t = blockIdx.x * 256 + threadIdx.x;
    if (t < n) d[t] = (_Float16)s[t];
}

__global__ void zero_f32_kernel(float* p, int n) {
    int t = blockIdx.x * 256 + threadIdx.x;
    if (t < n) p[t] = 0.0f;
}

// NCHW f32 -> [HW][C] f16 (channel-last activations for implicit GEMM)
__global__ void nchw_to_hwc_kernel(const float* __restrict__ s, _Float16* __restrict__ d,
                                   int C, int HW) {
    int t = blockIdx.x * 256 + threadIdx.x;
    if (t >= C * HW) return;
    int p = t / C, c = t - p * C;
    d[t] = (_Float16)s[c * HW + p];
}

// OIHW (O=256, I=Cin, 3x3) -> f16 [K=(t*Cin+ic)][N=o]
__global__ void repack3_kernel(const float* __restrict__ w, _Float16* __restrict__ d, int Cin) {
    int t = blockIdx.x * 256 + threadIdx.x;
    if (t >= Cin * 9 * 256) return;
    int k = t >> 8, o = t & 255;
    int tt = k / Cin, ic = k - tt * Cin;
    d[t] = (_Float16)w[(o * Cin + ic) * 9 + tt];
}

// OI (1x1) -> f16 [K=ic][N=o]
__global__ void repack1_kernel(const float* __restrict__ w, _Float16* __restrict__ d,
                               int Cin, int N) {
    int t = blockIdx.x * 256 + threadIdx.x;
    if (t >= Cin * N) return;
    int k = t / N, o = t - k * N;
    d[t] = (_Float16)w[o * Cin + k];
}

// fold BN: scale = g/sqrt(v+eps); bias = b - m*scale
__global__ void bn_fold_kernel(const float* g, const float* b, const float* m, const float* v,
                               float* sc, float* bi, int n) {
    int t = blockIdx.x * 256 + threadIdx.x;
    if (t >= n) return;
    float inv = g[t] * rsqrtf(v[t] + 1e-3f);
    sc[t] = inv;
    bi[t] = b[t] - m[t] * inv;
}

// box decode + per-anchor class max/argmax
__global__ void decode_kernel(const float* __restrict__ reg_raw, const float* __restrict__ obj_s,
                              const float* __restrict__ cls_s,
                              float* boxes, float* ccf, int* cid, float* scobj) {
    int a = blockIdx.x * 256 + threadIdx.x;
    if (a >= NT) return;
    int Wd, st, p;
    if (a < 15360)      { Wd = 160; st = 8;  p = a; }
    else if (a < 19200) { Wd = 80;  st = 16; p = a - 15360; }
    else                { Wd = 40;  st = 32; p = a - 19200; }
    float gx = (float)(p % Wd), gy = (float)(p / Wd);
    float cx = (reg_raw[a * 4 + 0] + gx) * (float)st;
    float cy = (reg_raw[a * 4 + 1] + gy) * (float)st;
    float bw = __expf(reg_raw[a * 4 + 2]) * (float)st;
    float bh = __expf(reg_raw[a * 4 + 3]) * (float)st;
    boxes[a * 4 + 0] = cx - 0.5f * bw;  boxes[a * 4 + 1] = cy - 0.5f * bh;
    boxes[a * 4 + 2] = cx + 0.5f * bw;  boxes[a * 4 + 3] = cy + 0.5f * bh;
    float best = -1e30f; int bi = 0;
    for (int j = 0; j < 30; j++) {
        float vv = cls_s[a * 30 + j];
        if (vv > best) { best = vv; bi = j; }
    }
    ccf[a] = best; cid[a] = bi; scobj[a] = obj_s[a];
}

// top-750 by obj (iterative argmax, stable ties -> smallest index)
__global__ void topk_kernel(float* scobj, int* sidx) {
    __shared__ float sv[1024];
    __shared__ int   si[1024];
    int tid = threadIdx.x;
    for (int t = 0; t < PRE; t++) {
        float best = -1e30f; int bi = 0x7fffffff;
        for (int j = tid; j < NT; j += 1024) {
            float v = scobj[j];
            if (v > best || (v == best && j < bi)) { best = v; bi = j; }
        }
        sv[tid] = best; si[tid] = bi; __syncthreads();
        for (int off = 512; off >= 1; off >>= 1) {
            if (tid < off) {
                if (sv[tid + off] > sv[tid] ||
                    (sv[tid + off] == sv[tid] && si[tid + off] < si[tid])) {
                    sv[tid] = sv[tid + off]; si[tid] = si[tid + off];
                }
            }
            __syncthreads();
        }
        if (tid == 0) { sidx[t] = si[0]; scobj[si[0]] = -1e30f; }
        __syncthreads();
    }
}

// stable descending sort of sc=obj*cls_conf via O(n^2) rank; build bo boxes
__global__ void sort_kernel(const int* sidx, const float* obj_s, const float* ccf,
                            const float* boxes, const int* cid,
                            int* sidx_ord, float* bo) {
    __shared__ float ssc[PRE];
    __shared__ int   ordr[PRE];
    __shared__ float red[1024];
    int t = threadIdx.x;
    int a = 0; float my = -1e30f;
    if (t < PRE) { a = sidx[t]; my = obj_s[a] * ccf[a]; ssc[t] = my; }
    __syncthreads();
    if (t < PRE) {
        int rank = 0;
        for (int j = 0; j < PRE; j++) {
            float sj = ssc[j];
            if (sj > my || (sj == my && j < t)) rank++;
        }
        ordr[rank] = t;
    }
    float lm = -1e30f;
    if (t < PRE)
        lm = fmaxf(fmaxf(boxes[a * 4 + 0], boxes[a * 4 + 1]),
                   fmaxf(boxes[a * 4 + 2], boxes[a * 4 + 3]));
    red[t] = lm; __syncthreads();
    for (int off = 512; off >= 1; off >>= 1) {
        if (t < off) red[t] = fmaxf(red[t], red[t + off]);
        __syncthreads();
    }
    float bm1 = red[0] + 1.0f;
    if (t < PRE) {
        int src = ordr[t]; int aa = sidx[src];
        sidx_ord[t] = aa;
        float of = (float)cid[aa] * bm1;
        bo[t * 4 + 0] = boxes[aa * 4 + 0] + of;
        bo[t * 4 + 1] = boxes[aa * 4 + 1] + of;
        bo[t * 4 + 2] = boxes[aa * 4 + 2] + of;
        bo[t * 4 + 3] = boxes[aa * 4 + 3] + of;
    }
}

// sequential-scan NMS (matches jax.lax.scan semantics) + select first 30
__global__ void nms_kernel(const float* __restrict__ bo, const int* __restrict__ sidx_ord,
                           int* fidx) {
    __shared__ float bx[PRE * 4];
    __shared__ float ar[PRE];
    __shared__ unsigned char kp[PRE];
    int t = threadIdx.x;
    for (int i = t; i < PRE; i += 1024) {
        float x0 = bo[i * 4 + 0], y0 = bo[i * 4 + 1];
        float x1 = bo[i * 4 + 2], y1 = bo[i * 4 + 3];
        bx[i * 4] = x0; bx[i * 4 + 1] = y0; bx[i * 4 + 2] = x1; bx[i * 4 + 3] = y1;
        ar[i] = fmaxf(x1 - x0, 0.f) * fmaxf(y1 - y0, 0.f);
        kp[i] = 1;
    }
    for (int i = 0; i < PRE; i++) {
        __syncthreads();
        if (!kp[i]) continue;
        float x0 = bx[i * 4], y0 = bx[i * 4 + 1], x1 = bx[i * 4 + 2], y1 = bx[i * 4 + 3];
        float ai = ar[i];
        for (int j = i + 1 + t; j < PRE; j += 1024) {
            float lx = fmaxf(x0, bx[j * 4]),     ly = fmaxf(y0, bx[j * 4 + 1]);
            float rx = fminf(x1, bx[j * 4 + 2]), ry = fminf(y1, bx[j * 4 + 3]);
            float inter = fmaxf(rx - lx, 0.f) * fmaxf(ry - ly, 0.f);
            float iou = inter / (ai + ar[j] - inter + 1e-9f);
            if (iou > 0.75f) kp[j] = 0;
        }
    }
    __syncthreads();
    if (t == 0) {
        int cnt = 0;
        for (int i = 0; i < PRE && cnt < 30; i++) if (kp[i])  fidx[cnt++] = sidx_ord[i];
        for (int i = 0; i < PRE && cnt < 30; i++) if (!kp[i]) fidx[cnt++] = sidx_ord[i];
    }
}

// gather 30 selected + 570 external tokens into f16 feature matrices + scores
__global__ void gather_kernel(const int* fidx,
                              const _Float16* __restrict__ cls_all,
                              const _Float16* __restrict__ reg_all,
                              const float* ocf, const float* orf,
                              const float* ocs, const float* ors,
                              const float* ccf, const float* obj_s,
                              _Float16* fcls, _Float16* freg, float* cs, float* fs) {
    int t = blockIdx.x * 256 + threadIdx.x;
    if (t >= NTOK * 256) return;
    int r = t >> 8, c = t & 255;
    if (r < 30) {
        int a = fidx[r];
        fcls[t] = cls_all[(long long)a * 256 + c];
        freg[t] = reg_all[(long long)a * 256 + c];
        if (c == 0) { cs[r] = ccf[a]; fs[r] = obj_s[a]; }
    } else {
        fcls[t] = (_Float16)ocf[(r - 30) * 256 + c];
        freg[t] = (_Float16)orf[(r - 30) * 256 + c];
        if (c == 0) { cs[r] = ocs[r - 30]; fs[r] = ors[r - 30]; }
    }
}

// per-64-lane-segment L2 norm (block of 256 = 4 heads x 64)
__device__ __forceinline__ float segnorm64(float v, float* red) {
    int t = threadIdx.x, dd = t & 63;
    red[t] = v * v;
    __syncthreads();
    for (int off = 32; off >= 1; off >>= 1) {
        if (dd < off) red[t] += red[t + off];
        __syncthreads();
    }
    float s = sqrtf(red[t - dd]);
    __syncthreads();
    return s;
}

// split QKV per head, normalize q/k (+v-normalized copy), write f16 (+transposed K)
__global__ void attn_prep_kernel(const float* __restrict__ qkvc, const float* __restrict__ qkvr,
                                 _Float16* qc, _Float16* kcT, _Float16* vc,
                                 _Float16* vcn, _Float16* vcnT,
                                 _Float16* qr, _Float16* krT) {
    __shared__ float red[256];
    int n = blockIdx.x, isreg = blockIdx.y, t = threadIdx.x;
    int h = t >> 6, dd = t & 63;
    const float* src = isreg ? qkvr : qkvc;
    float qv = src[n * 768 + t];
    float kv = src[n * 768 + 256 + t];
    float vv = src[n * 768 + 512 + t];
    float nq = segnorm64(qv, red);
    float nk = segnorm64(kv, red);
    _Float16* qd = isreg ? qr  : qc;
    _Float16* kd = isreg ? krT : kcT;
    qd[(h * NTOK + n) * 64 + dd] = (_Float16)(qv / nq);
    kd[(h * 64 + dd) * NTOK + n] = (_Float16)(kv / nk);
    if (!isreg) {
        float nv = segnorm64(vv, red);
        vc[(h * NTOK + n) * 64 + dd]    = (_Float16)vv;
        float vn = vv / nv;
        vcn[(h * NTOK + n) * 64 + dd]   = (_Float16)vn;
        vcnT[(h * 64 + dd) * NTOK + n]  = (_Float16)vn;
    }
}

__device__ __forceinline__ float blockReduce256(float v, float* red, int isSum) {
    int t = threadIdx.x;
    red[t] = v; __syncthreads();
    for (int off = 128; off >= 1; off >>= 1) {
        if (t < off) red[t] = isSum ? red[t] + red[t + off] : fmaxf(red[t], red[t + off]);
        __syncthreads();
    }
    float r = red[0]; __syncthreads();
    return r;
}

// fused: softmax(S*25)*score renorm for cls & reg branches, averaged attention,
// plus head-summed sim_attn accumulation
__global__ void attn_softmax_kernel(const float* __restrict__ Sc, const float* __restrict__ Sr,
                                    const float* cs, const float* fs,
                                    _Float16* attn_h, float* simattn) {
    __shared__ float ec[NTOK], er[NTOK], red[256];
    int q = blockIdx.x, h = blockIdx.y, t = threadIdx.x;
    const float* rc = Sc + (long long)h * NTOK * NTOK + (long long)q * NTOK;
    const float* rr = Sr + (long long)h * NTOK * NTOK + (long long)q * NTOK;
    float m1 = -1e30f, m2 = -1e30f;
    for (int k = t; k < NTOK; k += 256) { m1 = fmaxf(m1, rc[k]); m2 = fmaxf(m2, rr[k]); }
    m1 = blockReduce256(m1, red, 0);
    m2 = blockReduce256(m2, red, 0);
    float s1 = 0.f, s2 = 0.f;
    for (int k = t; k < NTOK; k += 256) {
        float e1 = __expf((rc[k] - m1) * 25.0f) * cs[k]; ec[k] = e1; s1 += e1;
        float e2 = __expf((rr[k] - m2) * 25.0f) * fs[k]; er[k] = e2; s2 += e2;
    }
    s1 = blockReduce256(s1, red, 1);
    s2 = blockReduce256(s2, red, 1);
    float i1 = 1.0f / s1, i2 = 1.0f / s2;
    for (int k = t; k < NTOK; k += 256) {
        float a = 0.5f * (ec[k] * i1 + er[k] * i2);
        attn_h[(long long)h * NTOK * NTOK + (long long)q * NTOK + k] = (_Float16)a;
        atomicAdd(&simattn[q * NTOK + k], 0.25f * a);
    }
}

// r2 = normalize( softmax(sim_attn) * (raw_sim > 0.75) )
__global__ void r2_kernel(const float* __restrict__ rawh, const float* __restrict__ simattn,
                          _Float16* r2) {
    __shared__ float w[NTOK], red[256];
    int q = blockIdx.x, t = threadIdx.x;
    float mx = -1e30f;
    for (int k = t; k < NTOK; k += 256) mx = fmaxf(mx, simattn[q * NTOK + k]);
    mx = blockReduce256(mx, red, 0);
    float s = 0.f;
    const int HH = NTOK * NTOK;
    for (int k = t; k < NTOK; k += 256) {
        int idx = q * NTOK + k;
        float raw = 0.25f * (rawh[idx] + rawh[HH + idx] + rawh[2 * HH + idx] + rawh[3 * HH + idx]);
        float e = (raw > 0.75f) ? __expf(simattn[idx] - mx) : 0.0f;
        w[k] = e; s += e;
    }
    s = blockReduce256(s, red, 1);
    float inv = 1.0f / s;
    for (int k = t; k < NTOK; k += 256) r2[q * NTOK + k] = (_Float16)(w[k] * inv);
}

// x_ori: vc [h][n][d] -> xcat[:, 256 + h*64 + d]
__global__ void copy_xori_kernel(const _Float16* __restrict__ vc, _Float16* __restrict__ xcat) {
    int t = blockIdx.x * 256 + threadIdx.x;
    if (t >= NTOK * 256) return;
    int n = t >> 8, c = t & 255, h = c >> 6, dd = c & 63;
    xcat[n * 512 + 256 + c] = vc[(h * NTOK + n) * 64 + dd];
}

__global__ void copy_f1_msa_kernel(const _Float16* __restrict__ f1, _Float16* __restrict__ msa) {
    int t = blockIdx.x * 256 + threadIdx.x;
    if (t >= NTOK * 512) return;
    int n = t >> 9, c = t & 511;
    msa[n * 1024 + 512 + c] = f1[t];
}

__global__ void final_copy_kernel(const float* __restrict__ fc, float* __restrict__ out) {
    int t = blockIdx.x * 256 + threadIdx.x;
    if (t >= 900) return;
    int r = t / 30, c = t - r * 30;
    out[t] = fc[r * 31 + c];   // drop column 30 (NUM_CLASSES slice)
}

// ===========================================================================
// Host orchestration
// ===========================================================================
extern "C" void kernel_launch(void* const* d_in, const int* in_sizes, int n_in,
                              void* d_out, int out_size, void* d_ws, size_t ws_size,
                              hipStream_t stream)
{
    (void)in_sizes; (void)n_in; (void)out_size; (void)ws_size;
    auto P = [&](int i) -> const float* { return (const float*)d_in[i]; };

    char* base = (char*)d_ws;
    size_t off = 0;
    auto alloc = [&](size_t bytes) -> void* {
        void* p = base + off;
        off = (off + bytes + 255) & ~(size_t)255;
        return p;
    };

    _Float16* xin   = (_Float16*)alloc((size_t)3932160 * 2);       // max-level input, [HW][Cin]
    _Float16* sbf   = (_Float16*)alloc((size_t)15360 * 256 * 2);   // stem out
    _Float16* tbf   = (_Float16*)alloc((size_t)15360 * 256 * 2);   // conv temp
    _Float16* cfb   = (_Float16*)alloc((size_t)15360 * 256 * 2);   // cls feat
    _Float16* rfall = (_Float16*)alloc((size_t)NT * 256 * 2);      // reg feats, all anchors
    _Float16* c2all = (_Float16*)alloc((size_t)NT * 256 * 2);      // cls2 feats, all anchors
    _Float16* wbuf  = (_Float16*)alloc((size_t)1048576 * 2);       // repacked weights
    float* bn_sc  = (float*)alloc(256 * 4);
    float* bn_bi  = (float*)alloc(256 * 4);
    float* reg_raw = (float*)alloc((size_t)NT * 4 * 4);
    float* obj_s   = (float*)alloc((size_t)NT * 4);
    float* cls_s   = (float*)alloc((size_t)NT * 30 * 4);
    float* boxes   = (float*)alloc((size_t)NT * 4 * 4);
    float* ccf     = (float*)alloc((size_t)NT * 4);
    int*   cid     = (int*)alloc((size_t)NT * 4);
    float* scobj   = (float*)alloc((size_t)NT * 4);
    int*   sidx    = (int*)alloc(752 * 4);
    int*   sidx_ord= (int*)alloc(752 * 4);
    float* bo      = (float*)alloc(3008 * 4);
    int*   fidx    = (int*)alloc(32 * 4);
    _Float16* fcls = (_Float16*)alloc((size_t)NTOK * 256 * 2);
    _Float16* freg = (_Float16*)alloc((size_t)NTOK * 256 * 2);
    float* cs      = (float*)alloc(NTOK * 4);
    float* fs      = (float*)alloc(NTOK * 4);
    float* qkvc    = (float*)alloc((size_t)NTOK * 768 * 4);
    float* qkvr    = (float*)alloc((size_t)NTOK * 768 * 4);
    _Float16* qc   = (_Float16*)alloc((size_t)4 * NTOK * 64 * 2);
    _Float16* kcT  = (_Float16*)alloc((size_t)4 * NTOK * 64 * 2);
    _Float16* vc   = (_Float16*)alloc((size_t)4 * NTOK * 64 * 2);
    _Float16* vcn  = (_Float16*)alloc((size_t)4 * NTOK * 64 * 2);
    _Float16* vcnT = (_Float16*)alloc((size_t)4 * NTOK * 64 * 2);
    _Float16* qr   = (_Float16*)alloc((size_t)4 * NTOK * 64 * 2);
    _Float16* krT  = (_Float16*)alloc((size_t)4 * NTOK * 64 * 2);
    float* Sc      = (float*)alloc((size_t)4 * NTOK * NTOK * 4);
    float* Sr      = (float*)alloc((size_t)4 * NTOK * NTOK * 4);
    float* rawh    = (float*)alloc((size_t)4 * NTOK * NTOK * 4);
    _Float16* attnh= (_Float16*)alloc((size_t)4 * NTOK * NTOK * 2);
    float* simattn = (float*)alloc((size_t)NTOK * NTOK * 4);
    _Float16* r2h  = (_Float16*)alloc((size_t)NTOK * NTOK * 2);
    _Float16* xcat = (_Float16*)alloc((size_t)NTOK * 512 * 2);
    _Float16* f1h  = (_Float16*)alloc((size_t)NTOK * 512 * 2);
    _Float16* msa  = (_Float16*)alloc((size_t)NTOK * 1024 * 2);
    _Float16* outb = (_Float16*)alloc((size_t)32 * 1024 * 2);
    float* fcbuf   = (float*)alloc(32 * 31 * 4);

    auto gemm = [&](const _Float16* A, int ldA, const _Float16* B,
                    float* oF, _Float16* oH, int ldC,
                    const float* scl, const float* bia, int act,
                    int M, int N, int K, int im2, int H, int W, int Cin,
                    long long bA, long long bB, long long bC, int nb) {
        dim3 g((unsigned)((M + 63) / 64), (unsigned)((N + 63) / 64), (unsigned)nb);
        gemm_wmma<<<g, dim3(128), 0, stream>>>(A, ldA, B, oF, oH, ldC, scl, bia, act,
                                               M, N, K, im2, H, W, Cin, bA, bB, bC);
    };
    auto grid1 = [&](int n) { return dim3((unsigned)((n + 255) / 256)); };

    struct Lvl { int H, W, HW, aoff, Cin; };
    const Lvl L[3] = {{96, 160, 15360, 0, 256},
                      {48,  80,  3840, 15360, 512},
                      {24,  40,   960, 19200, 1024}};

    // -------- per-level head convolutions (all via WMMA implicit GEMM) --------
    for (int l = 0; l < 3; l++) {
        const Lvl& v = L[l];
        const int pb = 7 + l * 41;
        nchw_to_hwc_kernel<<<grid1(v.HW * v.Cin), 256, 0, stream>>>(P(l), xin, v.Cin, v.HW);

        // stem: 1x1 Cin->256, BN + SiLU
        bn_fold_kernel<<<1, 256, 0, stream>>>(P(pb + 1), P(pb + 2), P(pb + 3), P(pb + 4),
                                              bn_sc, bn_bi, 256);
        repack1_kernel<<<grid1(v.Cin * 256), 256, 0, stream>>>(P(pb + 0), wbuf, v.Cin, 256);
        gemm(xin, v.Cin, wbuf, nullptr, sbf, 256, bn_sc, bn_bi, 1,
             v.HW, 256, v.Cin, 0, v.H, v.W, v.Cin, 0, 0, 0, 1);

        auto cbs3 = [&](int pw, const _Float16* in, _Float16* out) {
            bn_fold_kernel<<<1, 256, 0, stream>>>(P(pw + 1), P(pw + 2), P(pw + 3), P(pw + 4),
                                                  bn_sc, bn_bi, 256);
            repack3_kernel<<<grid1(2304 * 256), 256, 0, stream>>>(P(pw + 0), wbuf, 256);
            gemm(in, 256, wbuf, nullptr, out, 256, bn_sc, bn_bi, 1,
                 v.HW, 256, 2304, 1, v.H, v.W, 256, 0, 0, 0, 1);
        };
        _Float16* rf = rfall + (size_t)v.aoff * 256;
        _Float16* c2 = c2all + (size_t)v.aoff * 256;
        cbs3(pb + 25, sbf, tbf);  cbs3(pb + 30, tbf, rf);    // reg branch
        cbs3(pb + 5,  sbf, tbf);  cbs3(pb + 10, tbf, cfb);   // cls branch
        cbs3(pb + 15, sbf, tbf);  cbs3(pb + 20, tbf, c2);    // cls2 branch

        // 1x1 heads: obj (sigmoid), reg (raw), cls (sigmoid)
        repack1_kernel<<<grid1(256 * 1), 256, 0, stream>>>(P(pb + 39), wbuf, 256, 1);
        gemm(rf, 256, wbuf, obj_s + v.aoff, nullptr, 1, nullptr, P(pb + 40), 2,
             v.HW, 1, 256, 0, 0, 0, 0, 0, 0, 0, 1);
        repack1_kernel<<<grid1(256 * 4), 256, 0, stream>>>(P(pb + 37), wbuf, 256, 4);
        gemm(rf, 256, wbuf, reg_raw + (size_t)v.aoff * 4, nullptr, 4, nullptr, P(pb + 38), 3,
             v.HW, 4, 256, 0, 0, 0, 0, 0, 0, 0, 1);
        repack1_kernel<<<grid1(256 * 30), 256, 0, stream>>>(P(pb + 35), wbuf, 256, 30);
        gemm(cfb, 256, wbuf, cls_s + (size_t)v.aoff * 30, nullptr, 30, nullptr, P(pb + 36), 2,
             v.HW, 30, 256, 0, 0, 0, 0, 0, 0, 0, 1);
    }

    // -------- decode -> top-750 -> stable sort -> NMS -> gather 600 tokens ----
    decode_kernel<<<grid1(NT), 256, 0, stream>>>(reg_raw, obj_s, cls_s, boxes, ccf, cid, scobj);
    topk_kernel<<<1, 1024, 0, stream>>>(scobj, sidx);
    sort_kernel<<<1, 1024, 0, stream>>>(sidx, obj_s, ccf, boxes, cid, sidx_ord, bo);
    nms_kernel<<<1, 1024, 0, stream>>>(bo, sidx_ord, fidx);
    gather_kernel<<<grid1(NTOK * 256), 256, 0, stream>>>(fidx, c2all, rfall,
                                                         P(3), P(4), P(5), P(6),
                                                         ccf, obj_s, fcls, freg, cs, fs);

    // -------- attention block (all GEMMs via WMMA) -----------------------------
    f32_to_h_kernel<<<grid1(256 * 768), 256, 0, stream>>>(P(130), wbuf, 256 * 768);
    gemm(fcls, 256, wbuf, qkvc, nullptr, 768, nullptr, nullptr, 0,
         NTOK, 768, 256, 0, 0, 0, 0, 0, 0, 0, 1);
    f32_to_h_kernel<<<grid1(256 * 768), 256, 0, stream>>>(P(131), wbuf, 256 * 768);
    gemm(freg, 256, wbuf, qkvr, nullptr, 768, nullptr, nullptr, 0,
         NTOK, 768, 256, 0, 0, 0, 0, 0, 0, 0, 1);
    attn_prep_kernel<<<dim3(NTOK, 2), 256, 0, stream>>>(qkvc, qkvr, qc, kcT, vc, vcn, vcnT, qr, krT);

    const long long HB = (long long)NTOK * 64;           // per-head q/k/v stride
    const long long SB = (long long)NTOK * NTOK;         // per-head score stride
    gemm(qc,  64, kcT,  Sc,   nullptr, NTOK, nullptr, nullptr, 0, NTOK, NTOK, 64, 0,0,0,0, HB, HB, SB, 4);
    gemm(qr,  64, krT,  Sr,   nullptr, NTOK, nullptr, nullptr, 0, NTOK, NTOK, 64, 0,0,0,0, HB, HB, SB, 4);
    gemm(vcn, 64, vcnT, rawh, nullptr, NTOK, nullptr, nullptr, 0, NTOK, NTOK, 64, 0,0,0,0, HB, HB, SB, 4);

    zero_f32_kernel<<<grid1(NTOK * NTOK), 256, 0, stream>>>(simattn, NTOK * NTOK);
    attn_softmax_kernel<<<dim3(NTOK, 4), 256, 0, stream>>>(Sc, Sr, cs, fs, attnh, simattn);
    r2_kernel<<<NTOK, 256, 0, stream>>>(rawh, simattn, r2h);

    // xatt = attn @ v  -> xcat[:, h*64: (h+1)*64] ; x_ori -> xcat[:, 256:]
    gemm(attnh, NTOK, vc, nullptr, xcat, 512, nullptr, nullptr, 0,
         NTOK, 64, NTOK, 0, 0, 0, 0, SB, HB, 64, 4);
    copy_xori_kernel<<<grid1(NTOK * 256), 256, 0, stream>>>(vc, xcat);

    // f1 = xcat @ w1 + b1
    f32_to_h_kernel<<<grid1(512 * 512), 256, 0, stream>>>(P(132), wbuf, 512 * 512);
    gemm(xcat, 512, wbuf, nullptr, f1h, 512, nullptr, P(133), 3,
         NTOK, 512, 512, 0, 0, 0, 0, 0, 0, 0, 1);

    // msa = concat(r2 @ f1, f1)
    copy_f1_msa_kernel<<<grid1(NTOK * 512), 256, 0, stream>>>(f1h, msa);
    gemm(r2h, NTOK, f1h, nullptr, msa, 1024, nullptr, nullptr, 0,
         NTOK, 512, NTOK, 0, 0, 0, 0, 0, 0, 0, 1);

    // out[:30] = msa[:30] @ w2 + b2 ; fc = out @ wp + bp
    f32_to_h_kernel<<<grid1(1024 * 1024), 256, 0, stream>>>(P(134), wbuf, 1024 * 1024);
    gemm(msa, 1024, wbuf, nullptr, outb, 1024, nullptr, P(135), 3,
         30, 1024, 1024, 0, 0, 0, 0, 0, 0, 0, 1);
    f32_to_h_kernel<<<grid1(1024 * 31), 256, 0, stream>>>(P(136), wbuf, 1024 * 31);
    gemm(outb, 1024, wbuf, fcbuf, nullptr, 31, nullptr, P(137), 3,
         30, 31, 1024, 0, 0, 0, 0, 0, 0, 0, 1);

    final_copy_kernel<<<grid1(900), 256, 0, stream>>>(fcbuf, (float*)d_out);
}
